// QModel_21878563406029
// MI455X (gfx1250) — matrix operand order repaired and checked by reading, not compiled, not run on hardware
//
#include <hip/hip_runtime.h>
#include <hip/hip_bf16.h>
#include <math.h>

// ---------------------------------------------------------------------------
// Types for WMMA fragments (gfx1250, wave32)
// ---------------------------------------------------------------------------
typedef __attribute__((ext_vector_type(16))) __bf16 v16bf;
typedef __attribute__((ext_vector_type(8)))  float  v8f;

struct alignas(16) U128 { unsigned x, y, z, w; };
struct alignas(16) F128 { float x, y, z, w; };

union Frag {
    v16bf  bf;
    U128   q[2];
    unsigned u[8];
};

__device__ __forceinline__ unsigned short f32_to_bf16(float f) {
    unsigned u = __float_as_uint(f);
    u += 0x7FFFu + ((u >> 16) & 1u);      // round-to-nearest-even
    return (unsigned short)(u >> 16);
}
__device__ __forceinline__ unsigned pack_bf16x2(float lo, float hi) {
    return (unsigned)f32_to_bf16(lo) | ((unsigned)f32_to_bf16(hi) << 16);
}
__device__ __forceinline__ float bf16_to_f32(unsigned short h) {
    return __uint_as_float(((unsigned)h) << 16);
}

__device__ __constant__ float c_sigma[8] = {
    0.5515909f, 1.8886297f, 1.3225029f, 1.2316629f,
    2.1884933f, 1.7750372f, 1.3677907f, 1.3820058f
};

// ---------------------------------------------------------------------------
// gfx1250 async global->LDS copy (ASYNCcnt) with safe fallback.
// Builtin signature (from hipcc diagnostic): pointers to
// int __attribute__((vector_size(16))) in AS(1) (global) / AS(3) (LDS).
// ---------------------------------------------------------------------------
#if defined(__has_builtin)
#  if __has_builtin(__builtin_amdgcn_global_load_async_to_lds_b128)
#    define HAVE_ASYNC_LDS 1
#  endif
#  if __has_builtin(__builtin_amdgcn_s_wait_asynccnt)
#    define HAVE_WAIT_ASYNC_BUILTIN 1
#  endif
#endif

typedef int v4i_t __attribute__((vector_size(16)));
typedef __attribute__((address_space(1))) v4i_t as1_v4i;
typedef __attribute__((address_space(3))) v4i_t as3_v4i;

// copy 64 bytes (one column's K-chunk) global -> LDS
__device__ __forceinline__ void async_copy64(const unsigned short* gp,
                                             unsigned short* lp) {
#if defined(HAVE_ASYNC_LDS)
    as1_v4i* g = (as1_v4i*)(v4i_t*)(void*)const_cast<unsigned short*>(gp);
    as3_v4i* l = (as3_v4i*)(v4i_t*)(void*)lp;
    __builtin_amdgcn_global_load_async_to_lds_b128(g, l, 0,  0);
    __builtin_amdgcn_global_load_async_to_lds_b128(g, l, 16, 0);
    __builtin_amdgcn_global_load_async_to_lds_b128(g, l, 32, 0);
    __builtin_amdgcn_global_load_async_to_lds_b128(g, l, 48, 0);
#else
    U128 r0 = *reinterpret_cast<const U128*>(gp);
    U128 r1 = *reinterpret_cast<const U128*>(gp + 8);
    U128 r2 = *reinterpret_cast<const U128*>(gp + 16);
    U128 r3 = *reinterpret_cast<const U128*>(gp + 24);
    *reinterpret_cast<U128*>(lp)      = r0;
    *reinterpret_cast<U128*>(lp + 8)  = r1;
    *reinterpret_cast<U128*>(lp + 16) = r2;
    *reinterpret_cast<U128*>(lp + 24) = r3;
#endif
}

template <int NOUT>
__device__ __forceinline__ void wait_async() {
#if defined(HAVE_ASYNC_LDS)
#  if defined(HAVE_WAIT_ASYNC_BUILTIN)
    __builtin_amdgcn_s_wait_asynccnt(NOUT);
#  else
    asm volatile("s_wait_asynccnt %0" :: "n"(NOUT) : "memory");
#  endif
#endif
}

// ---------------------------------------------------------------------------
// Weight transpose + f32->bf16 convert:  W(K,N) row-major -> WT(N,Kp) bf16,
// zero-padded in K (layer0: 1008 -> 1024).
// ---------------------------------------------------------------------------
__global__ __launch_bounds__(256) void transpose_cvt(
    const float* __restrict__ W, unsigned short* __restrict__ WT,
    int K, int N, int Kp)
{
    int idx = blockIdx.x * 256 + threadIdx.x;
    if (idx >= N * Kp) return;
    int n = idx / Kp;
    int k = idx - n * Kp;
    WT[idx] = (k < K) ? f32_to_bf16(W[(long)k * N + n]) : (unsigned short)0;
}

// ---------------------------------------------------------------------------
// Fused GEMM + bias + CELU(alpha=0.1), bf16 WMMA (16x16x32), f32 accum.
//   A  : (M, lda) f32 (AF32, in-register bf16 convert) or bf16 row-major
//   BT : (N, Kp) bf16 row-major (B transposed -> contiguous K per column)
//   Out: (M, N) bf16 row-major
// Block = 256 threads = 8 waves = 4 row-groups x 2 col-groups; each wave owns
// 16 rows x NTW*16 columns. B k-panels staged in LDS (TRIPLE-buffered, async
// prefetch 2 panels ahead; ASYNCcnt thresholds 8/4/0 drain only the consumed
// panel). Panel layout is fragment order (col-major, 32 contiguous K = 64B per
// column) so one B fragment is two ds_load_b128; B fragments are prefetched
// 2 tiles deep through three named Frags (constant indices -> no scratch).
// A fragments ping-pong through two named locals via a k-loop unrolled by 2
// (nk is even for all layers: 32, 8, 6).
// ---------------------------------------------------------------------------
template <int NTW, bool AF32>
__global__ __launch_bounds__(256) void gemm_celu(
    const void* __restrict__ Av, const unsigned short* __restrict__ BT,
    const float* __restrict__ bias, unsigned short* __restrict__ Out,
    int N, int Kp, int Kvalid, int lda)
{
    __shared__ alignas(16) unsigned short ldsB[3][8192];   // 3 x 16KB panels

    const int tid    = threadIdx.x;
    const int lane   = tid & 31;
    const int wave   = tid >> 5;
    const int rowgrp = wave >> 1;            // 0..3
    const int colgrp = wave & 1;             // 0..1
    const int half   = lane >> 4;            // 0: lanes 0-15, 1: lanes 16-31
    const int l16    = lane & 15;
    const long rowbase = ((long)blockIdx.x * 4 + rowgrp) * 16;
    const long arow    = rowbase + l16;      // A row held by this lane
    const int colbase  = colgrp * NTW * 16;  // this wave's first output column
    const int nk = Kp / 32;                  // even for all layers (32, 8, 6)

    v8f acc[NTW] = {};

    // ---- helpers -----------------------------------------------------------
    auto prefetchB = [&](int ks) {           // one thread per column
        if (tid < N) {
            const unsigned short* gp = BT + (long)tid * Kp + ks * 32;
            unsigned short* lp = &ldsB[ks % 3][tid * 32];
            async_copy64(gp, lp);
        }
    };
    auto loadA = [&](Frag& a, int ks) {
        const int c0 = ks * 32 + half * 16;
        const int c1 = c0 + 8;
        if constexpr (AF32) {
            const float* ap = (const float*)Av + arow * (long)lda;
            F128 f0, f1, f2, f3;
            if (c0 + 8 <= Kvalid) {
                f0 = *reinterpret_cast<const F128*>(ap + c0);
                f1 = *reinterpret_cast<const F128*>(ap + c0 + 4);
            } else { f0 = F128{}; f1 = F128{}; }
            if (c1 + 8 <= Kvalid) {
                f2 = *reinterpret_cast<const F128*>(ap + c1);
                f3 = *reinterpret_cast<const F128*>(ap + c1 + 4);
            } else { f2 = F128{}; f3 = F128{}; }
            a.u[0] = pack_bf16x2(f0.x, f0.y);
            a.u[1] = pack_bf16x2(f0.z, f0.w);
            a.u[2] = pack_bf16x2(f1.x, f1.y);
            a.u[3] = pack_bf16x2(f1.z, f1.w);
            a.u[4] = pack_bf16x2(f2.x, f2.y);
            a.u[5] = pack_bf16x2(f2.z, f2.w);
            a.u[6] = pack_bf16x2(f3.x, f3.y);
            a.u[7] = pack_bf16x2(f3.z, f3.w);
        } else {
            const unsigned short* ap = (const unsigned short*)Av + arow * (long)lda;
            a.q[0] = *reinterpret_cast<const U128*>(ap + c0);
            a.q[1] = *reinterpret_cast<const U128*>(ap + c1);
        }
    };
    auto loadBfrag = [&](Frag& b, int buf, int t) {
        const unsigned short* lp =
            &ldsB[buf][(colbase + t * 16 + l16) * 32 + half * 16];
        b.q[0] = *reinterpret_cast<const U128*>(lp);
        b.q[1] = *reinterpret_cast<const U128*>(lp + 8);
    };
    auto mma_step = [&](Frag& a, int buf) {  // B fragments prefetched 2 deep
        Frag b0, b1, b2;
        loadBfrag(b0, buf, 0);
        loadBfrag(b1, buf, 1);               // NTW >= 2 always (5, 6, 8)
#pragma unroll
        for (int t = 0; t < NTW; ++t) {
            if (t + 2 < NTW) {
                const int w = (t + 2) % 3;   // constant under unroll
                if      (w == 0) loadBfrag(b0, buf, t + 2);
                else if (w == 1) loadBfrag(b1, buf, t + 2);
                else             loadBfrag(b2, buf, t + 2);
            }
            const int r = t % 3;
            acc[t] = __builtin_amdgcn_wmma_f32_16x16x32_bf16(
                false, a.bf, false, (r == 0 ? b0.bf : (r == 1 ? b1.bf : b2.bf)),
                (short)0, acc[t], false, false);
        }
    };

    // ---- software-pipelined main loop --------------------------------------
    Frag a0, a1;
    prefetchB(0);
    prefetchB(1);
    loadA(a0, 0);

    for (int ks = 0; ks < nk; ks += 2) {
        // even step: consume panel ks (a0); panels ks+1, ks+2 stay in flight
        if (ks + 2 < nk) prefetchB(ks + 2);
        loadA(a1, ks + 1);
        if (ks + 2 < nk) wait_async<8>(); else wait_async<4>();
        __syncthreads();
        mma_step(a0, ks % 3);
        __syncthreads();

        // odd step: consume panel ks+1 (a1); panels ks+2, ks+3 stay in flight
        if (ks + 3 < nk) prefetchB(ks + 3);
        if (ks + 2 < nk) loadA(a0, ks + 2);
        if      (ks + 3 < nk) wait_async<8>();
        else if (ks + 2 < nk) wait_async<4>();
        else                  wait_async<0>();
        __syncthreads();
        mma_step(a1, (ks + 1) % 3);
        __syncthreads();
    }

    // ---- epilogue: bias + CELU(0.1), store bf16 ----------------------------
#pragma unroll
    for (int t = 0; t < NTW; ++t) {
        const int col = colbase + t * 16 + l16;
        const float bv = bias[col];
#pragma unroll
        for (int r = 0; r < 8; ++r) {
            float x = acc[t][r] + bv;
            float y = x > 0.f ? x : 0.1f * (expf(x * 10.f) - 1.f);
            const long row = rowbase + half * 8 + r;  // VGPR r -> M = r + 8*half
            Out[row * (long)N + col] = f32_to_bf16(y);
        }
    }
}

// ---------------------------------------------------------------------------
// Last layer (160 -> 1) + softplus + species mask -> chi (f32 per atom)
// ---------------------------------------------------------------------------
__global__ __launch_bounds__(256) void chi_kernel(
    const unsigned short* __restrict__ H, const float* __restrict__ w3,
    const float* __restrict__ b3, const int* __restrict__ species,
    float* __restrict__ chi, int M, int K)
{
    int a = blockIdx.x * 256 + threadIdx.x;
    if (a >= M) return;
    const unsigned short* hp = H + (long)a * K;
    float s = b3[0];
    for (int k = 0; k < K; ++k)
        s = fmaf(bf16_to_f32(hp[k]), w3[k], s);
    float sp = (s > 20.f) ? s : log1pf(expf(s));
    chi[a] = (species[a] != -1) ? sp : 0.f;
}

// ---------------------------------------------------------------------------
// Pairwise charge transfer + residual redistribution.
// One block = one molecule (64 atoms, 2 waves). The erf/Coulomb off-diagonal
// is dead code in the reference (only diagonal j_ii = 1/(sigma*sqrt(pi)) is
// consumed), so it is skipped.
// ---------------------------------------------------------------------------
__global__ __launch_bounds__(64) void charge_kernel(
    const int* __restrict__ species, const float* __restrict__ coords,
    const float* __restrict__ netq, const float* __restrict__ chi,
    float* __restrict__ out_q)
{
    __shared__ float sx[64], sy[64], sz[64], ssig[64], schi[64];
    __shared__ int   smask[64];
    __shared__ float s1[64], s2[64];

    const int b = blockIdx.x;
    const int i = threadIdx.x;
    const int sp = species[b * 64 + i];
    const int m  = (sp != -1);
    const float sig = c_sigma[sp & 7];     // -1 wraps to last entry
    const float* cp = coords + ((long)b * 64 + i) * 3;
    const float xi = cp[0], yi = cp[1], zi = cp[2];
    const float chi_i = chi[b * 64 + i];

    sx[i] = xi; sy[i] = yi; sz[i] = zi;
    ssig[i] = sig; schi[i] = chi_i; smask[i] = m;
    __syncthreads();

    const float INV_A0      = 1.8897259885789233f;  // 1 / 0.529177249
    const float INV_SQRT_PI = 0.5641895835477563f;
    const float jii_i = INV_SQRT_PI / sig;          // 1/(sigma*sqrt(pi))

    float q = 0.f;
    for (int j = 0; j < 64; ++j) {
        float dx = xi - sx[j], dy = yi - sy[j], dz = zi - sz[j];
        float d  = sqrtf(dx * dx + dy * dy + dz * dz + 1e-16f) * INV_A0;
        float sj = ssig[j];
        float R  = d - (sig + sj);
        float decay = expf(-R) * (1.f + R + R * R * (1.f / 3.f));
        float jii_j = INV_SQRT_PI / sj;
        float t = decay * jii_i * jii_j * (schi[j] - chi_i);
        q += (m && smask[j] && (j != i)) ? t : 0.f;
    }

    s1[i] = q;
    s2[i] = q * q;
    __syncthreads();
    for (int off = 32; off > 0; off >>= 1) {
        if (i < off) { s1[i] += s1[i + off]; s2[i] += s2[i + off]; }
        __syncthreads();
    }
    const float sumq  = s1[0];
    const float sumq2 = fmaxf(s2[0], 1e-8f);
    const float pred  = q + (netq[b] - sumq) * (q * q) / sumq2;
    out_q[b * 64 + i] = pred;
}

__global__ __launch_bounds__(256) void copy_species(
    const int* __restrict__ sp, int* __restrict__ out, int n)
{
    int a = blockIdx.x * 256 + threadIdx.x;
    if (a < n) out[a] = sp[a];
}

// ---------------------------------------------------------------------------
// Launch
// ---------------------------------------------------------------------------
extern "C" void kernel_launch(void* const* d_in, const int* in_sizes, int n_in,
                              void* d_out, int out_size, void* d_ws, size_t ws_size,
                              hipStream_t stream)
{
    const int*   species = (const int*)  d_in[0];   // (512, 64)
    const float* coords  = (const float*)d_in[1];   // (512, 64, 3)
    const float* netq    = (const float*)d_in[2];   // (512,)
    const float* aev     = (const float*)d_in[3];   // (512, 64, 1008)
    const float* w0 = (const float*)d_in[4];        // (1008, 256)
    const float* b0 = (const float*)d_in[5];
    const float* w1 = (const float*)d_in[6];        // (256, 192)
    const float* b1 = (const float*)d_in[7];
    const float* w2 = (const float*)d_in[8];        // (192, 160)
    const float* b2 = (const float*)d_in[9];
    const float* w3 = (const float*)d_in[10];       // (160, 1)
    const float* b3 = (const float*)d_in[11];

    const int M = 512 * 64;                         // 32768 atom rows

    // workspace layout (all 16B aligned)
    char* ws = (char*)d_ws;
    unsigned short* w0T = (unsigned short*)(ws);              // 256*1024*2  = 524288
    unsigned short* w1T = (unsigned short*)(ws + 524288);     // 192*256*2   = 98304
    unsigned short* w2T = (unsigned short*)(ws + 622592);     // 160*192*2   = 61440
    unsigned short* h0  = (unsigned short*)(ws + 684032);     // 32768*256*2 = 16777216
    unsigned short* h1  = (unsigned short*)(ws + 17461248);   // 32768*192*2 = 12582912
    unsigned short* h2  = (unsigned short*)(ws + 30044160);   // 32768*160*2 = 10485760
    float*          chi = (float*)         (ws + 40529920);   // 32768*4

    // weight transpose + bf16 convert (B as N x Kp so panels are contiguous K)
    transpose_cvt<<<(256 * 1024 + 255) / 256, 256, 0, stream>>>(w0, w0T, 1008, 256, 1024);
    transpose_cvt<<<(192 * 256  + 255) / 256, 256, 0, stream>>>(w1, w1T, 256,  192, 256);
    transpose_cvt<<<(160 * 192  + 255) / 256, 256, 0, stream>>>(w2, w2T, 192,  160, 192);

    // MLP: 3 fused GEMM+CELU layers on the WMMA pipes
    // block covers 64 rows x all N; wave = 16 rows x N/2 cols
    gemm_celu<8, true ><<<M / 64, 256, 0, stream>>>(aev, w0T, b0, h0, 256, 1024, 1008, 1008);
    gemm_celu<6, false><<<M / 64, 256, 0, stream>>>(h0,  w1T, b1, h1, 192, 256,  256,  256);
    gemm_celu<5, false><<<M / 64, 256, 0, stream>>>(h1,  w2T, b2, h2, 160, 192,  192,  192);

    // final projection + softplus + mask
    chi_kernel<<<M / 256, 256, 0, stream>>>(h2, w3, b3, species, chi, M, 160);

    // outputs: species (int32) then pred_charges (f32)
    int*   out_sp = (int*)d_out;
    float* out_q  = (float*)d_out + M;
    copy_species<<<M / 256, 256, 0, stream>>>(species, out_sp, M);
    charge_kernel<<<512, 64, 0, stream>>>(species, coords, netq, chi, out_q);
}
